// ColorHistograms_81277961109920
// MI455X (gfx1250) — matrix operand order repaired and checked by reading, not compiled, not run on hardware
//
#include <hip/hip_runtime.h>
#include <hip/hip_bf16.h>

typedef __attribute__((ext_vector_type(16))) _Float16 v16h;
typedef __attribute__((ext_vector_type(8)))  _Float16 v8h;
typedef __attribute__((ext_vector_type(8)))  float    v8f;
typedef __attribute__((ext_vector_type(4)))  int      v4i;

#define T_FRAMES 1024
#define NBINS    512
#define HW       4096
#define KPAD     128
#define LOOKUP   101
#define HALF     50

// ---------------------------------------------------------------------------
// Kernel 0: convert W [128 x 101] fp32 -> Wp [128 x 128] f16, zero-padded K.
// ---------------------------------------------------------------------------
__global__ void convW_kernel(const float* __restrict__ W, _Float16* __restrict__ Wp) {
  int idx = blockIdx.x * 256 + threadIdx.x;   // 128*128 = 16384 threads
  int o = idx >> 7;
  int k = idx & 127;
  Wp[idx] = (k < LOOKUP) ? (_Float16)W[o * LOOKUP + k] : (_Float16)0.0f;
}

// ---------------------------------------------------------------------------
// Kernel 1: per-frame 512-bin color histogram + L2 normalize -> f16.
// One 256-thread block per frame (8192 blocks). LDS atomics (ds_add_u32),
// int4 coalesced loads (global_load_b128), wave32 shuffle reduction.
// ---------------------------------------------------------------------------
__global__ void hist_kernel(const int* __restrict__ frames,
                            _Float16* __restrict__ histF16) {
  __shared__ unsigned sh[NBINS];
  __shared__ float red[8];

  const int f   = blockIdx.x;            // frame id = b*T + t
  const int b   = f >> 10;
  const int t   = f & (T_FRAMES - 1);
  const int tid = threadIdx.x;           // 256 threads

  sh[tid]       = 0u;
  sh[tid + 256] = 0u;
  __syncthreads();

  const size_t plane = (size_t)T_FRAMES * HW;               // one channel of one batch
  const int* base = frames + (size_t)b * 3 * plane + (size_t)t * HW;

  #pragma unroll
  for (int i = 0; i < 4; ++i) {
    const int off = (i * 256 + tid) * 4;                    // 4096 pixels / block
    v4i r  = *(const v4i*)(base + off);
    v4i g  = *(const v4i*)(base + plane + off);
    v4i bl = *(const v4i*)(base + 2 * plane + off);
    #pragma unroll
    for (int j = 0; j < 4; ++j) {
      unsigned bin = (((unsigned)r[j]  >> 5) << 6) |
                     (((unsigned)g[j]  >> 5) << 3) |
                      ((unsigned)bl[j] >> 5);
      atomicAdd(&sh[bin], 1u);                              // ds_add_u32
    }
  }
  __syncthreads();

  const unsigned c0 = sh[tid];
  const unsigned c1 = sh[tid + 256];
  float ss = (float)c0 * (float)c0 + (float)c1 * (float)c1;

  // wave32 reduction, then cross-wave via LDS
  #pragma unroll
  for (int o = 16; o > 0; o >>= 1) ss += __shfl_down(ss, o, 32);
  if ((tid & 31) == 0) red[tid >> 5] = ss;
  __syncthreads();

  float tot = 0.0f;
  #pragma unroll
  for (int i = 0; i < 8; ++i) tot += red[i];                // uniform, no divergence
  const float inv = rsqrtf(tot);                            // tot >= 4096^2/512 > 0

  _Float16* dst = histF16 + (size_t)f * NBINS;
  dst[tid]       = (_Float16)((float)c0 * inv);
  dst[tid + 256] = (_Float16)((float)c1 * inv);
}

// ---------------------------------------------------------------------------
// Fragment loaders (16-bit A 16x32, 16-bit B 32x16), wave32, contiguous k = 16*(lane/16) + s.
// ---------------------------------------------------------------------------
__device__ __forceinline__ v16h load_a_frag(const _Float16* row) {
  // row points at this lane's A-row (m = lane&15) + kb + khA (khA = (lane>>4)*16)
  v8h lo = *(const v8h*)(row);        // K = khA + 0..7
  v8h hi = *(const v8h*)(row + 8);    // K = khA + 8..15
  v16h a;
  #pragma unroll
  for (int i = 0; i < 8; ++i) { a[i] = lo[i]; a[i + 8] = hi[i]; }
  return a;
}

__device__ __forceinline__ v16h load_b_frag(const _Float16* col) {
  // col points at this lane's B-column row (n = lane&15) + kb + khB (khB = (lane>>4)*16)
  v8h b0 = *(const v8h*)(col);
  v8h b1 = *(const v8h*)(col + 8);
  v16h bf;
  #pragma unroll
  for (int i = 0; i < 8; ++i) { bf[i] = b0[i]; bf[i + 8] = b1[i]; }
  return bf;
}

// ---------------------------------------------------------------------------
// Banded similarity tile body. EDGE=false: all 9 column tiles are guaranteed
// in-range (rt in [4,60)) -> guard-free inner loop. EDGE=true: wave-uniform
// range guards (tiles are 16-aligned), zero fragments off-range.
// B fragments are processed in chunks of 3 (6 clause-able b128 loads, then
// 3 WMMAs): enough load/WMMA overlap without exceeding the 256-VGPR window
// (acc 72 + A 8 + B 24 + addressing ~ 120 VGPRs, spill-free).
// ---------------------------------------------------------------------------
template <bool EDGE>
__device__ __forceinline__ void simband_tile(const _Float16* __restrict__ hist,
                                             _Float16* __restrict__ win,
                                             int b, int t0, int lane) {
  const int m   = lane & 15;
  const int khA = (lane >> 4) << 4;       // 0 or 16
  const int n   = lane & 15;
  const int khB = (lane >> 4) << 4;       // 0 or 16

  const _Float16* Abase = hist + (size_t)(b * T_FRAMES + t0 + m) * NBINS;
  // this lane's B-column row for tile j: hist row (b*T + t0 + (j-4)*16 + n)
  const _Float16* Bbase = hist + (size_t)(b * T_FRAMES + t0 - 64 + n) * NBINS + khB;

  v8f acc[9] = {};

  for (int kb = 0; kb < NBINS; kb += 32) {
    if (kb + 32 < NBINS)
      __builtin_prefetch(Abase + kb + 32 + khA, 0, 3);      // near-cache prefetch
    const v16h a = load_a_frag(Abase + kb + khA);

    #pragma unroll
    for (int jj = 0; jj < 9; jj += 3) {
      // chunk of 3 B fragments: issue all 6 b128 loads, then 3 WMMAs
      v16h bfr[3];
      #pragma unroll
      for (int u = 0; u < 3; ++u) {
        const int j = jj + u;
        if (EDGE) {
          const int s0 = t0 + (j - 4) * 16;    // 16-aligned -> uniform guard
          bfr[u] = (v16h){};
          if (s0 >= 0 && s0 < T_FRAMES)
            bfr[u] = load_b_frag(Bbase + (size_t)j * 16 * NBINS + kb);
        } else {
          bfr[u] = load_b_frag(Bbase + (size_t)j * 16 * NBINS + kb);
        }
      }
      #pragma unroll
      for (int u = 0; u < 3; ++u)
        acc[jj + u] = __builtin_amdgcn_wmma_f32_16x16x32_f16(
                          false, a, false, bfr[u], (short)0, acc[jj + u],
                          false, false);
    }
  }

  // Scatter band into padded window buffer (unique writer per (t,l))
  #pragma unroll
  for (int j = 0; j < 9; ++j) {
    const int s0 = t0 + (j - 4) * 16;
    if (EDGE && (s0 < 0 || s0 >= T_FRAMES)) continue;
    const int s = s0 + n;
    #pragma unroll
    for (int i = 0; i < 8; ++i) {
      const int mm = i + ((lane >> 4) << 3);   // C/D layout: VGPR i, lane half
      const int t  = t0 + mm;
      const int l  = s - t + HALF;
      if (l >= 0 && l < LOOKUP)
        win[(size_t)(b * T_FRAMES + t) * KPAD + l] = (_Float16)acc[j][i];
    }
  }
}

// ---------------------------------------------------------------------------
// Kernel 2: banded similarity. One wave per (batch, 16-row tile).
// 9 column tiles of 16 cover the +/-50 window; K=512 in 16 WMMA steps.
// Block-uniform interior/edge split: 56/64 row tiles take the guard-free path.
// ---------------------------------------------------------------------------
__global__ void simband_kernel(const _Float16* __restrict__ hist,
                               _Float16* __restrict__ win) {
  const int blk  = blockIdx.x;            // B * T/16 = 512
  const int b    = blk >> 6;
  const int rt   = blk & 63;
  const int t0   = rt * 16;
  const int lane = threadIdx.x;           // 32 threads = 1 wave

  if (rt >= 4 && rt < 60) {
    simband_tile<false>(hist, win, b, t0, lane);   // interior: no range checks
  } else {
    simband_tile<true>(hist, win, b, t0, lane);    // edge: guarded B fragments
  }
}

// ---------------------------------------------------------------------------
// Kernel 3: out = relu(win @ Wp^T + bias). [8192 x 128] x [128 x 128] f16 WMMA.
// One wave per 16x16 output tile: (8192/16) * (128/16) = 4096 waves.
// All 4 fragment pairs preloaded (64 VGPRs), then 4 chained WMMAs.
// ---------------------------------------------------------------------------
__global__ void linear_kernel(const _Float16* __restrict__ win,
                              const _Float16* __restrict__ Wp,
                              const float* __restrict__ bias,
                              float* __restrict__ out) {
  const int blk  = blockIdx.x;
  const int nt   = blk & 7;
  const int mt   = blk >> 3;
  const int lane = threadIdx.x;

  const int m   = lane & 15;
  const int khA = (lane >> 4) << 4;
  const int n   = lane & 15;
  const int khB = (lane >> 4) << 4;

  const _Float16* Abase = win + (size_t)(mt * 16 + m) * KPAD;
  const _Float16* Bbase = Wp  + (size_t)(nt * 16 + n) * KPAD;

  // Preload the whole K=128 strip (4 A-frags + 4 B-frags = 16 b128 loads)
  v16h afr[4], bfr[4];
  #pragma unroll
  for (int q = 0; q < 4; ++q) {
    afr[q] = load_a_frag(Abase + q * 32 + khA);
    bfr[q] = load_b_frag(Bbase + q * 32 + khB);
  }

  v8f acc = {};
  #pragma unroll
  for (int q = 0; q < 4; ++q)
    acc = __builtin_amdgcn_wmma_f32_16x16x32_f16(
              false, afr[q], false, bfr[q], (short)0, acc, false, false);

  const float bv = bias[nt * 16 + n];
  #pragma unroll
  for (int i = 0; i < 8; ++i) {
    const int mm = i + ((lane >> 4) << 3);
    const float v = acc[i] + bv;
    out[(size_t)(mt * 16 + mm) * 128 + nt * 16 + n] = v > 0.0f ? v : 0.0f;
  }
}

// ---------------------------------------------------------------------------
// Launch
// ---------------------------------------------------------------------------
extern "C" void kernel_launch(void* const* d_in, const int* in_sizes, int n_in,
                              void* d_out, int out_size, void* d_ws, size_t ws_size,
                              hipStream_t stream) {
  (void)in_sizes; (void)n_in; (void)out_size; (void)ws_size;

  const int*   frames = (const int*)d_in[0];     // [8,3,1024,64,64] int32
  const float* W      = (const float*)d_in[1];   // [128,101] f32
  const float* bias   = (const float*)d_in[2];   // [128] f32
  float*       out    = (float*)d_out;           // [8,1024,128] f32

  char* ws = (char*)d_ws;
  const size_t histBytes = (size_t)8192 * NBINS * sizeof(_Float16);  // 8 MB
  const size_t winBytes  = (size_t)8192 * KPAD  * sizeof(_Float16);  // 2 MB
  _Float16* histF16 = (_Float16*)ws;
  _Float16* win     = (_Float16*)(ws + histBytes);
  _Float16* Wp      = (_Float16*)(ws + histBytes + winBytes);

  // 0) weight conversion / padding
  convW_kernel<<<64, 256, 0, stream>>>(W, Wp);

  // 1) per-frame histograms (memory-bound sweep of the 400 MB input)
  hist_kernel<<<8192, 256, 0, stream>>>(frames, histF16);

  // 2) zero padded window buffer (reference zero-pads the band edges)
  hipMemsetAsync(win, 0, winBytes, stream);

  // 3) banded similarity via f16 WMMA (L2-resident, ~1.2 GFLOP instead of 8.6)
  simband_kernel<<<8 * (T_FRAMES / 16), 32, 0, stream>>>(histF16, win);

  // 4) linear(101->128) + ReLU via f16 WMMA
  linear_kernel<<<(8192 / 16) * (128 / 16), 32, 0, stream>>>(win, Wp, bias, out);
}
